// TransformerBlock_29016799052237
// MI455X (gfx1250) — matrix operand order repaired and checked by reading, not compiled, not run on hardware
//
#include <hip/hip_runtime.h>
#include <stdint.h>

#define BATCH   4
#define SEQ     4096
#define DIM     1024
#define HEADS   16
#define DHEAD   64
#define KPROJ   64
#define MLPD    4096
#define MROWS   (BATCH * SEQ)      // 16384
#define EPSF    1e-5f

typedef unsigned short u16;
typedef __bf16 bf16x16 __attribute__((ext_vector_type(16)));
typedef float  f32x8   __attribute__((ext_vector_type(8)));
typedef unsigned int u32x4 __attribute__((ext_vector_type(4)));

// ---------- scalar helpers ----------
__device__ __forceinline__ u16 f2bf(float f) {
  union { float f; unsigned int u; } c; c.f = f;
  unsigned int u = c.u;
  return (u16)((u + 0x7FFFu + ((u >> 16) & 1u)) >> 16);   // RNE
}
__device__ __forceinline__ float bf2f(u16 h) {
  union { unsigned int u; float f; } c; c.u = ((unsigned int)h) << 16;
  return c.f;
}
__device__ __forceinline__ float wred_sum(float v) {
#pragma unroll
  for (int o = 16; o > 0; o >>= 1) v += __shfl_xor(v, o, 32);
  return v;
}
__device__ __forceinline__ float wred_max(float v) {
#pragma unroll
  for (int o = 16; o > 0; o >>= 1) v = fmaxf(v, __shfl_xor(v, o, 32));
  return v;
}

// ---------- weight transpose + cast:  W[K,N] f32  ->  Wt[N,K] bf16 ----------
__global__ __launch_bounds__(256) void transpose_cast_kernel(
    const float* __restrict__ in, u16* __restrict__ out, int K, int N) {
  __shared__ float tile[32][33];
  const int k0 = blockIdx.y * 32, n0 = blockIdx.x * 32;
  const int tx = threadIdx.x, ty = threadIdx.y;      // 32 x 8
#pragma unroll
  for (int j = 0; j < 4; ++j)
    tile[ty + j * 8][tx] = in[(size_t)(k0 + ty + j * 8) * N + n0 + tx];
  __syncthreads();
#pragma unroll
  for (int j = 0; j < 4; ++j)
    out[(size_t)(n0 + ty + j * 8) * K + k0 + tx] = f2bf(tile[tx][ty + j * 8]);
}

// ---------- LayerNorm row kernel: f32 in, bf16 out ----------
__global__ __launch_bounds__(256) void layernorm_kernel(
    const float* __restrict__ x, const float* __restrict__ g,
    const float* __restrict__ b, u16* __restrict__ out) {
  const int row = blockIdx.x;
  const int t = threadIdx.x;
  const float* xr = x + (size_t)row * DIM;
  float v[4], s = 0.f, s2 = 0.f;
#pragma unroll
  for (int i = 0; i < 4; ++i) { v[i] = xr[t + i * 256]; s += v[i]; s2 += v[i] * v[i]; }
  s = wred_sum(s); s2 = wred_sum(s2);
  __shared__ float red[16];
  const int w = t >> 5, lane = t & 31;
  if (lane == 0) { red[w] = s; red[8 + w] = s2; }
  __syncthreads();
  float ts = 0.f, t2 = 0.f;
#pragma unroll
  for (int j = 0; j < 8; ++j) { ts += red[j]; t2 += red[8 + j]; }
  const float mu = ts * (1.0f / DIM);
  const float var = t2 * (1.0f / DIM) - mu * mu;
  const float rstd = rsqrtf(var + EPSF);
  u16* orow = out + (size_t)row * DIM;
#pragma unroll
  for (int i = 0; i < 4; ++i) {
    const int c = t + i * 256;
    orow[c] = f2bf((v[i] - mu) * rstd * g[c] + b[c]);
  }
}

// ---------- column sums ----------
__global__ __launch_bounds__(256) void colsum_bf16_kernel(
    const u16* __restrict__ in, float* __restrict__ out,
    int batch, int rows, int cols) {
  const int idx = blockIdx.x * 256 + threadIdx.x;
  if (idx >= batch * cols) return;
  const int b = idx / cols, c = idx % cols;
  const u16* p = in + (size_t)b * rows * cols + c;
  float acc = 0.f;
  for (int r = 0; r < rows; ++r) acc += bf2f(p[(size_t)r * cols]);
  out[idx] = acc;
}
__global__ __launch_bounds__(256) void colsum_f32_kernel(
    const float* __restrict__ in, float* __restrict__ out, int rows, int cols) {
  const int c = blockIdx.x * 256 + threadIdx.x;
  if (c >= cols) return;
  float acc = 0.f;
  for (int r = 0; r < rows; ++r) acc += in[(size_t)r * cols + c];
  out[c] = acc;
}

// ---------- rank-1 Linformer attention core: one wave per (b,h,s) ----------
__global__ __launch_bounds__(256) void linformer_attn_kernel(
    const u16* __restrict__ q, const float* __restrict__ ksum,
    const float* __restrict__ vsum, const float* __restrict__ Esum,
    const float* __restrict__ Fsum, u16* __restrict__ out) {
  const int wg = blockIdx.x * 8 + (threadIdx.x >> 5);
  const int lane = threadIdx.x & 31;
  const int s = wg % SEQ;
  const int h = (wg / SEQ) % HEADS;
  const int b = wg / (SEQ * HEADS);
  const size_t qbase = ((size_t)(b * SEQ + s)) * (HEADS * DHEAD) + h * DHEAD;
  const int kbase = (b * HEADS + h) * DHEAD;
  const int d0 = lane * 2;
  // t = scale * (q . ksum)
  float t = bf2f(q[qbase + d0]) * ksum[kbase + d0] +
            bf2f(q[qbase + d0 + 1]) * ksum[kbase + d0 + 1];
  t = wred_sum(t) * 0.125f;                      // DH^-0.5
  // softmax over 64 projected keys: logits = t * Esum
  const float l0 = t * Esum[d0], l1 = t * Esum[d0 + 1];
  const float m = wred_max(fmaxf(l0, l1));
  const float p0 = __expf(l0 - m), p1 = __expf(l1 - m);
  const float denom = wred_sum(p0 + p1);
  const float num   = wred_sum(p0 * Fsum[d0] + p1 * Fsum[d0 + 1]);
  const float coeff = num / denom;
  out[qbase + d0]     = f2bf(coeff * vsum[kbase + d0]);
  out[qbase + d0 + 1] = f2bf(coeff * vsum[kbase + d0 + 1]);
}

// ---------- bf16 WMMA GEMM:  C[M,N] = A[M,K] * Bt[N,K]^T  (+bias, gelu, res) ----------
// Block tile 256x128, 8 waves in 4x2, 64x64 per wave (16 WMMA per K-step per
// wave; 16 ds_load_b128 feed 16 v_wmma -> 1.0 LDS reads per matrix op).
union FragAB { bf16x16 v; u32x4 u[2]; };

template <bool BIAS, bool GELU, bool RES, bool OUT_BF16>
__global__ __launch_bounds__(256) void gemm_bf16_wmma(
    const u16* __restrict__ A, const u16* __restrict__ Bt,
    const float* __restrict__ bias, const float* __restrict__ res,
    void* __restrict__ outp, int M, int N, int K) {
  // Rows padded to 40 shorts (80B): stride-20-dword pattern puts the 16
  // fragment-reading lanes on 16 distinct LDS banks.
  __shared__ u16 Abuf[256 * 40];
  __shared__ u16 Bbuf[128 * 40];
  const int t = threadIdx.x;
  const int lane = t & 31, w = t >> 5;
  const int wrow = (w >> 1) * 64;      // 4 wave-rows of 64
  const int wcol = (w & 1) * 64;       // 2 wave-cols of 64
  const int m0 = blockIdx.y * 256;
  const int n0 = blockIdx.x * 128;
  const int lrow = lane & 15;
  const int lhalf = lane >> 4;

  f32x8 acc[4][4] = {};

  for (int kt = 0; kt < K; kt += 32) {
    // cooperative stage: A 256x32, Bt 128x32 (both K-major, bf16)
#pragma unroll
    for (int i = 0; i < 4; ++i) {
      const int c = t + i * 256;       // 1024 chunks of 8 bf16
      const int row = c >> 2;
      const int c8 = (c & 3) * 8;
      *(u32x4*)&Abuf[row * 40 + c8] =
          *(const u32x4*)(A + (size_t)(m0 + row) * K + kt + c8);
    }
#pragma unroll
    for (int i = 0; i < 2; ++i) {
      const int c = t + i * 256;       // 512 chunks of 8 bf16
      const int row = c >> 2;
      const int c8 = (c & 3) * 8;
      *(u32x4*)&Bbuf[row * 40 + c8] =
          *(const u32x4*)(Bt + (size_t)(n0 + row) * K + kt + c8);
    }
    if (kt + 32 < K) {  // hint the next tiles toward L2 (global_prefetch_b8)
      __builtin_prefetch(A + (size_t)(m0 + t) * K + kt + 32, 0, 1);
      __builtin_prefetch(Bt + (size_t)(n0 + (t >> 1)) * K + kt + 32, 0, 1);
    }
    __syncthreads();

    FragAB a[4], bb[4];
#pragma unroll
    for (int tn = 0; tn < 4; ++tn) {
      // B 32x16 bf16 layout: lanes 0-15 K=0..15, lanes 16-31 K=16..31 (N = lane&15)
      const u16* pb = &Bbuf[(wcol + tn * 16 + lrow) * 40 + lhalf * 16];
      bb[tn].u[0] = *(const u32x4*)pb;
      bb[tn].u[1] = *(const u32x4*)(pb + 8);
    }
#pragma unroll
    for (int tm = 0; tm < 4; ++tm) {
      // A 16x32 bf16 layout: lanes 0-15 K=0..15, lanes 16-31 K=16..31
      const u16* pa = &Abuf[(wrow + tm * 16 + lrow) * 40 + lhalf * 16];
      a[tm].u[0] = *(const u32x4*)pa;
      a[tm].u[1] = *(const u32x4*)(pa + 8);
    }
#pragma unroll
    for (int tm = 0; tm < 4; ++tm)
#pragma unroll
      for (int tn = 0; tn < 4; ++tn)
        acc[tm][tn] = __builtin_amdgcn_wmma_f32_16x16x32_bf16(
            false, a[tm].v, false, bb[tn].v, (short)0, acc[tm][tn], false, false);
    __syncthreads();
  }

  // epilogue
  float biasv[4];
  if (BIAS) {
#pragma unroll
    for (int tn = 0; tn < 4; ++tn) biasv[tn] = bias[n0 + wcol + tn * 16 + lrow];
  }
  float* outf = (float*)outp;
  u16* outb = (u16*)outp;
#pragma unroll
  for (int tm = 0; tm < 4; ++tm) {
#pragma unroll
    for (int r = 0; r < 8; ++r) {
      const int m = m0 + wrow + tm * 16 + lhalf * 8 + r;   // C/D: M = r + half*8
#pragma unroll
      for (int tn = 0; tn < 4; ++tn) {
        const int n = n0 + wcol + tn * 16 + lrow;          // C/D: N = lane&15
        float v = acc[tm][tn][r];
        if (BIAS) v += biasv[tn];
        if (GELU) v = 0.5f * v * (1.0f + erff(v * 0.70710678118f));
        if (RES)  v += res[(size_t)m * N + n];
        if (OUT_BF16) outb[(size_t)m * N + n] = f2bf(v);
        else          outf[(size_t)m * N + n] = v;
      }
    }
  }
}

// ---------- driver ----------
extern "C" void kernel_launch(void* const* d_in, const int* in_sizes, int n_in,
                              void* d_out, int out_size, void* d_ws, size_t ws_size,
                              hipStream_t stream) {
  (void)in_sizes; (void)n_in; (void)out_size; (void)ws_size;
  const float* x    = (const float*)d_in[0];
  const float* Wq   = (const float*)d_in[1];
  const float* Wk   = (const float*)d_in[2];
  const float* Wv   = (const float*)d_in[3];
  const float* E    = (const float*)d_in[4];
  const float* Fm   = (const float*)d_in[5];
  const float* Wo   = (const float*)d_in[6];
  const float* bo   = (const float*)d_in[7];
  const float* ln1g = (const float*)d_in[8];
  const float* ln1b = (const float*)d_in[9];
  const float* ln2g = (const float*)d_in[10];
  const float* ln2b = (const float*)d_in[11];
  const float* W1   = (const float*)d_in[12];
  const float* b1   = (const float*)d_in[13];
  const float* W2   = (const float*)d_in[14];
  const float* b2   = (const float*)d_in[15];
  float* out = (float*)d_out;

  char* base = (char*)d_ws;
  size_t off = 0;
  auto wsalloc = [&](size_t bytes) -> void* {
    void* p = base + off;
    off += (bytes + 255) & ~(size_t)255;
    return p;
  };
  // qbuf..attnbuf are contiguous (128 MiB) and reused as hbuf later.
  u16*   xnbuf   = (u16*)wsalloc((size_t)MROWS * DIM * 2);    // LN1 out, reused as LN2 out
  u16*   qbuf    = (u16*)wsalloc((size_t)MROWS * DIM * 2);    // q
  u16*   kbuf    = (u16*)wsalloc((size_t)MROWS * DIM * 2);    // k
  u16*   vbuf    = (u16*)wsalloc((size_t)MROWS * DIM * 2);    // v
  u16*   attnbuf = (u16*)wsalloc((size_t)MROWS * DIM * 2);    // attention output
  float* x1buf   = (float*)wsalloc((size_t)MROWS * DIM * 4);  // residual stream
  u16*   WqT     = (u16*)wsalloc((size_t)DIM * DIM * 2);
  u16*   WkT     = (u16*)wsalloc((size_t)DIM * DIM * 2);
  u16*   WvT     = (u16*)wsalloc((size_t)DIM * DIM * 2);
  u16*   WoT     = (u16*)wsalloc((size_t)DIM * DIM * 2);
  u16*   W1T     = (u16*)wsalloc((size_t)DIM * MLPD * 2);
  u16*   W2T     = (u16*)wsalloc((size_t)MLPD * DIM * 2);
  float* ksumbuf = (float*)wsalloc((size_t)BATCH * DIM * 4);
  float* vsumbuf = (float*)wsalloc((size_t)BATCH * DIM * 4);
  float* EsumBuf = (float*)wsalloc((size_t)KPROJ * 4);
  float* FsumBuf = (float*)wsalloc((size_t)KPROJ * 4);
  u16*   hbuf    = qbuf;    // q/k/v/attn dead once h is produced
  u16*   xn2buf  = xnbuf;   // xn dead after q/k/v GEMMs

  const dim3 tb(32, 8);
  // weight casts/transposes (once per launch; amortized over 64 M-tiles)
  transpose_cast_kernel<<<dim3(DIM / 32, DIM / 32), tb, 0, stream>>>(Wq, WqT, DIM, DIM);
  transpose_cast_kernel<<<dim3(DIM / 32, DIM / 32), tb, 0, stream>>>(Wk, WkT, DIM, DIM);
  transpose_cast_kernel<<<dim3(DIM / 32, DIM / 32), tb, 0, stream>>>(Wv, WvT, DIM, DIM);
  transpose_cast_kernel<<<dim3(DIM / 32, DIM / 32), tb, 0, stream>>>(Wo, WoT, DIM, DIM);
  transpose_cast_kernel<<<dim3(MLPD / 32, DIM / 32), tb, 0, stream>>>(W1, W1T, DIM, MLPD);
  transpose_cast_kernel<<<dim3(DIM / 32, MLPD / 32), tb, 0, stream>>>(W2, W2T, MLPD, DIM);

  colsum_f32_kernel<<<1, 256, 0, stream>>>(E, EsumBuf, SEQ, KPROJ);
  colsum_f32_kernel<<<1, 256, 0, stream>>>(Fm, FsumBuf, SEQ, KPROJ);

  // LN1 -> xn (bf16)
  layernorm_kernel<<<MROWS, 256, 0, stream>>>(x, ln1g, ln1b, xnbuf);

  // q/k/v = xn @ W{q,k,v}
  const dim3 gD(DIM / 128, MROWS / 256);
  gemm_bf16_wmma<false, false, false, true><<<gD, 256, 0, stream>>>(xnbuf, WqT, nullptr, nullptr, qbuf, MROWS, DIM, DIM);
  gemm_bf16_wmma<false, false, false, true><<<gD, 256, 0, stream>>>(xnbuf, WkT, nullptr, nullptr, kbuf, MROWS, DIM, DIM);
  gemm_bf16_wmma<false, false, false, true><<<gD, 256, 0, stream>>>(xnbuf, WvT, nullptr, nullptr, vbuf, MROWS, DIM, DIM);

  // rank-1 attention: column sums + per-(b,h,s) scalar softmax
  colsum_bf16_kernel<<<(BATCH * DIM + 255) / 256, 256, 0, stream>>>(kbuf, ksumbuf, BATCH, SEQ, DIM);
  colsum_bf16_kernel<<<(BATCH * DIM + 255) / 256, 256, 0, stream>>>(vbuf, vsumbuf, BATCH, SEQ, DIM);
  linformer_attn_kernel<<<(BATCH * HEADS * SEQ) / 8, 256, 0, stream>>>(qbuf, ksumbuf, vsumbuf, EsumBuf, FsumBuf, attnbuf);

  // x1 = x + attn @ Wo + bo
  gemm_bf16_wmma<true, false, true, false><<<gD, 256, 0, stream>>>(attnbuf, WoT, bo, x, x1buf, MROWS, DIM, DIM);

  // LN2 -> xn2 (bf16)
  layernorm_kernel<<<MROWS, 256, 0, stream>>>(x1buf, ln2g, ln2b, xn2buf);

  // h = gelu(xn2 @ W1 + b1)
  gemm_bf16_wmma<true, true, false, true><<<dim3(MLPD / 128, MROWS / 256), 256, 0, stream>>>(
      xn2buf, W1T, b1, nullptr, hbuf, MROWS, MLPD, DIM);

  // out = x1 + h @ W2 + b2
  gemm_bf16_wmma<true, false, true, false><<<gD, 256, 0, stream>>>(
      hbuf, W2T, b2, x1buf, out, MROWS, DIM, MLPD);
}
